// AdvancedCOMPER_9088150798646
// MI455X (gfx1250) — matrix operand order repaired and checked by reading, not compiled, not run on hardware
//
#include <hip/hip_runtime.h>
#include <hip/hip_bf16.h>

#define DEV_INLINE __device__ __forceinline__

typedef _Float16 v16h __attribute__((ext_vector_type(16)));
typedef _Float16 v8h  __attribute__((ext_vector_type(8)));
typedef float    v8f  __attribute__((ext_vector_type(8)));

namespace {
constexpr int kP   = 100000;   // paths
constexpr int kL   = 4;        // steps
constexpr int kIN  = 192;      // 3*EMB
constexpr int kNB2 = 256;      // reduction blocks
constexpr int kChunk = (kP + kNB2 - 1) / kNB2;   // 391

// workspace layout (float offsets)
constexpr size_t OFF_SCORE = 0;                               // kP (padded 100352)
constexpr size_t OFF_PEMB  = 100352;                          // kP*64
constexpr size_t OFF_CVEC  = OFF_PEMB + (size_t)kP * 64;      // 128
constexpr size_t OFF_PMAX  = OFF_CVEC + 128;                  // kNB2
constexpr size_t OFF_PSUM  = OFF_PMAX + kNB2;                 // kNB2
constexpr size_t OFF_PAGG  = OFF_PSUM + kNB2;                 // kNB2*64
constexpr size_t OFF_FINAL = OFF_PAGG + (size_t)kNB2 * 64;    // [0]=max [1]=sum [2]=1/sum
}

DEV_INLINE float sigmoidf(float x) { return 1.0f / (1.0f + __expf(-x)); }

// A fragment (16x32 f16): lane<16 holds row=lane, K={kb..kb+7, kb+16..kb+23};
// lane>=16 holds row=lane-16, K={kb+8..kb+15, kb+24..kb+31}.
DEV_INLINE v16h load_A(const _Float16* base, int lane, int kb, int rs) {
  const int row = lane & 15;
  const int sub = (lane >> 4) << 3;            // 0 or 8
  const _Float16* p = base + row * rs + kb + sub;
  v8h lo = *(const v8h*)p;
  v8h hi = *(const v8h*)(p + 16);
  return __builtin_shufflevector(lo, hi, 0,1,2,3,4,5,6,7,8,9,10,11,12,13,14,15);
}

// B fragment (32x16 f16) from row-major weights W[n][k] (row stride rs):
// lane<16 -> col n0+lane, K={kb..kb+7, kb+16..kb+23}; lane>=16 -> col n0+lane-16, K={kb+8..kb+15, kb+24..kb+31}.
DEV_INLINE v16h load_B(const _Float16* w, int lane, int n0, int kb, int rs) {
  const int col = n0 + (lane & 15);
  const int ks  = kb + ((lane >> 4) << 3);     // 0 or 8
  const _Float16* p = w + col * rs + ks;
  v8h lo = *(const v8h*)p;
  v8h hi = *(const v8h*)(p + 16);
  return __builtin_shufflevector(lo, hi, 0,1,2,3,4,5,6,7,8,9,10,11,12,13,14,15);
}

DEV_INLINE v8f wmma_f16(v16h a, v16h b, v8f c) {
  return __builtin_amdgcn_wmma_f32_16x16x32_f16(false, a, false, b, (short)0, c, false, false);
}

// ---------------- K0: ctx vector for attention --------------------------
__global__ void k0_ctx(const int* user_idx, const int* item_idx,
                       const float* user_emb, const float* item_emb,
                       const float* att_W1, const float* att_b1, float* cvec) {
  __shared__ float ctxS[128];
  int t = threadIdx.x;                         // 128 threads
  int uid = user_idx[0], iid = item_idx[0];
  ctxS[t] = (t < 64) ? user_emb[(size_t)uid * 64 + t]
                     : item_emb[(size_t)iid * 64 + (t - 64)];
  __syncthreads();
  float a = att_b1[t];
  for (int k = 0; k < 128; ++k) a += att_W1[(size_t)t * 192 + 64 + k] * ctxS[k];
  cvec[t] = a;
}

// ---------------- K1: gather + LSTM + attention scores (WMMA) -----------
__global__ __launch_bounds__(256)
void k1_lstm_att(const int* node_ids, const int* node_types, const int* rel_idx,
                 const float* user_emb, const float* item_emb,
                 const float* rel_emb, const float* type_emb,
                 const float* W_ih, const float* W_hh,
                 const float* b_ih, const float* b_hh,
                 const float* att_W1, const float* att_W2, const float* att_b2,
                 const float* cvecG, float* scoreG, float* pembG) {
  extern __shared__ char smem[];
  _Float16* sWih = (_Float16*)smem;            // 256*192
  _Float16* sWhh = sWih + 256 * 192;           // 256*64
  _Float16* sAw1 = sWhh + 256 * 64;            // 128*64 (att_W1[:, :64])
  _Float16* sX   = sAw1 + 128 * 64;            // 8 waves * 16*192
  _Float16* sH   = sX + 8 * 16 * 192;          // 8 waves * 16*64
  float* sBias   = (float*)(sH + 8 * 16 * 64); // 256
  float* sCvec   = sBias + 256;                // 128

  const int tid  = threadIdx.x;
  const int w    = tid >> 5;
  const int lane = tid & 31;

  for (int i = tid; i < 256 * 192; i += 256) sWih[i] = (_Float16)W_ih[i];
  for (int i = tid; i < 256 * 64;  i += 256) sWhh[i] = (_Float16)W_hh[i];
  for (int i = tid; i < 128 * 64;  i += 256) {
    int n = i >> 6, k = i & 63;
    sAw1[i] = (_Float16)att_W1[(size_t)n * 192 + k];
  }
  if (tid < 256) sBias[tid] = b_ih[tid] + b_hh[tid];
  if (tid < 128) sCvec[tid] = cvecG[tid];
  __syncthreads();

  _Float16* sXw = sX + w * (16 * 192);
  _Float16* sHw = sH + w * (16 * 64);
  const int pbase = blockIdx.x * 128 + w * 16;

  v8f cs[4];
  #pragma unroll
  for (int dt = 0; dt < 4; ++dt)
    #pragma unroll
    for (int r = 0; r < 8; ++r) cs[dt][r] = 0.0f;

  for (int t = 0; t < kL; ++t) {
    // stage x_t for this wave's 16 paths: [pi][k] f16 (gather, fp32 -> f16)
    for (int idx = lane; idx < 16 * 192; idx += 32) {
      int pi = idx / 192, k = idx - pi * 192;
      int p = pbase + pi; if (p >= kP) p = kP - 1;
      int nid = node_ids[(size_t)p * kL + t];
      int nty = node_types[(size_t)p * kL + t];
      int rl  = rel_idx[(size_t)p * kL + t];
      float v;
      if (k < 64)       v = (nty == 0) ? user_emb[(size_t)nid * 64 + k]
                                       : item_emb[(size_t)nid * 64 + k];
      else if (k < 128) v = type_emb[(size_t)nty * 64 + (k - 64)];
      else              v = rel_emb[(size_t)rl * 64 + (k - 128)];
      sXw[idx] = (_Float16)v;
    }
    asm volatile("s_wait_dscnt 0" ::: "memory");

    // z = x W_ih^T + h W_hh^T + b   (16 paths x 256 gates, f32 accum)
    v8f acc[16];
    #pragma unroll
    for (int nt = 0; nt < 16; ++nt) {
      float bv = sBias[nt * 16 + (lane & 15)];
      #pragma unroll
      for (int r = 0; r < 8; ++r) acc[nt][r] = bv;
    }
    #pragma unroll
    for (int kc = 0; kc < 6; ++kc) {
      v16h a = load_A(sXw, lane, kc * 32, 192);
      #pragma unroll
      for (int nt = 0; nt < 16; ++nt)
        acc[nt] = wmma_f16(a, load_B(sWih, lane, nt * 16, kc * 32, 192), acc[nt]);
    }
    if (t > 0) {
      #pragma unroll
      for (int kc = 0; kc < 2; ++kc) {
        v16h a = load_A(sHw, lane, kc * 32, 64);
        #pragma unroll
        for (int nt = 0; nt < 16; ++nt)
          acc[nt] = wmma_f16(a, load_B(sWhh, lane, nt * 16, kc * 32, 64), acc[nt]);
      }
    }

    // gates (i,f,g,o) -> c,h ; write h back to LDS in A-layout source form
    #pragma unroll
    for (int dt = 0; dt < 4; ++dt) {
      #pragma unroll
      for (int r = 0; r < 8; ++r) {
        float ig = sigmoidf(acc[dt][r]);
        float fg = sigmoidf(acc[4 + dt][r]);
        float gg = tanhf(acc[8 + dt][r]);
        float og = sigmoidf(acc[12 + dt][r]);
        float c  = fg * cs[dt][r] + ig * gg;
        cs[dt][r] = c;
        float hv = og * tanhf(c);
        int M = r + ((lane >> 4) << 3);
        int d = dt * 16 + (lane & 15);
        sHw[M * 64 + d] = (_Float16)hv;
      }
    }
    asm volatile("s_wait_dscnt 0" ::: "memory");
  }

  // attention: relu(h W1a^T + cvec) . W2 + b2
  v8f aacc[8];
  #pragma unroll
  for (int nt = 0; nt < 8; ++nt) {
    float bv = sCvec[nt * 16 + (lane & 15)];
    #pragma unroll
    for (int r = 0; r < 8; ++r) aacc[nt][r] = bv;
  }
  #pragma unroll
  for (int kc = 0; kc < 2; ++kc) {
    v16h a = load_A(sHw, lane, kc * 32, 64);
    #pragma unroll
    for (int nt = 0; nt < 8; ++nt)
      aacc[nt] = wmma_f16(a, load_B(sAw1, lane, nt * 16, kc * 32, 64), aacc[nt]);
  }
  float part[8];
  #pragma unroll
  for (int r = 0; r < 8; ++r) part[r] = 0.0f;
  #pragma unroll
  for (int nt = 0; nt < 8; ++nt) {
    float w2 = att_W2[nt * 16 + (lane & 15)];
    #pragma unroll
    for (int r = 0; r < 8; ++r) part[r] += fmaxf(aacc[nt][r], 0.0f) * w2;
  }
  float ab2 = att_b2[0];
  #pragma unroll
  for (int r = 0; r < 8; ++r) {
    float s = part[r];
    s += __shfl_xor(s, 1, 32);
    s += __shfl_xor(s, 2, 32);
    s += __shfl_xor(s, 4, 32);
    s += __shfl_xor(s, 8, 32);
    if ((lane & 15) == 0) {
      int p = pbase + r + ((lane >> 4) << 3);
      if (p < kP) scoreG[p] = s + ab2;
    }
  }

  // path_embs (h_n) to workspace, f32
  for (int idx = lane; idx < 16 * 64; idx += 32) {
    int pi = idx >> 6, d = idx & 63;
    int p = pbase + pi;
    if (p < kP) pembG[(size_t)p * 64 + d] = (float)sHw[idx];
  }
}

// ---------------- K2/K3: softmax max reduction --------------------------
__global__ void k2_blockmax(const float* sc, float* pmax) {
  __shared__ float red[256];
  int b = blockIdx.x, t = threadIdx.x;
  int beg = b * kChunk, end = beg + kChunk; if (end > kP) end = kP;
  float m = -3.4e38f;
  for (int i = beg + t; i < end; i += 256) m = fmaxf(m, sc[i]);
  red[t] = m; __syncthreads();
  for (int o = 128; o > 0; o >>= 1) {
    if (t < o) red[t] = fmaxf(red[t], red[t + o]);
    __syncthreads();
  }
  if (t == 0) pmax[b] = red[0];
}

__global__ void k3_finalmax(const float* pmax, float* finals) {
  __shared__ float red[256];
  int t = threadIdx.x;
  red[t] = pmax[t]; __syncthreads();
  for (int o = 128; o > 0; o >>= 1) {
    if (t < o) red[t] = fmaxf(red[t], red[t + o]);
    __syncthreads();
  }
  if (t == 0) finals[0] = red[0];
}

// ---------------- K4: exp + partial sum + partial aggregation -----------
__global__ void k4_expagg(const float* sc, const float* pemb, const float* finals,
                          float* outw, float* psum, float* pagg) {
  __shared__ float aggL[4 * 64];
  __shared__ float sumL[4];
  int b = blockIdx.x, t = threadIdx.x;
  int beg = b * kChunk, end = beg + kChunk; if (end > kP) end = kP;
  float m = finals[0];
  for (int i = beg + t; i < end; i += 256) outw[i] = __expf(sc[i] - m);
  int g = t >> 6, d = t & 63;
  float a = 0.0f, es = 0.0f;
  for (int i = beg + g; i < end; i += 4) {
    float e = __expf(sc[i] - m);
    a += e * pemb[(size_t)i * 64 + d];
    if (d == 0) es += e;
  }
  aggL[g * 64 + d] = a;
  if (d == 0) sumL[g] = es;
  __syncthreads();
  if (t < 64) pagg[(size_t)b * 64 + t] =
      aggL[t] + aggL[64 + t] + aggL[128 + t] + aggL[192 + t];
  if (t == 0) psum[b] = sumL[0] + sumL[1] + sumL[2] + sumL[3];
}

// ---------------- K5: final sums + dueling heads + safety ---------------
__global__ void k5_heads(const float* psum, const float* pagg, float* finals,
                         const int* user_idx, const int* item_idx,
                         const float* user_emb, const float* item_emb,
                         const float* val_W1, const float* val_b1,
                         const float* val_W2, const float* val_b2,
                         const float* adv_W1, const float* adv_b1,
                         const float* adv_W2, const float* adv_b2,
                         const float* saf_W1, const float* saf_b1,
                         const float* saf_W2, const float* saf_b2,
                         float* d_out) {
  __shared__ float red[256];
  __shared__ float aggS[64];
  __shared__ float ctxS[128];
  __shared__ float hidV[32], hidA[32], hidS[32];
  int t = threadIdx.x;
  red[t] = psum[t]; __syncthreads();
  for (int o = 128; o > 0; o >>= 1) {
    if (t < o) red[t] += red[t + o];
    __syncthreads();
  }
  float S = red[0];
  float invS = 1.0f / S;
  if (t == 0) { finals[1] = S; finals[2] = invS; }
  if (t < 64) {
    float a = 0.0f;
    for (int b = 0; b < kNB2; ++b) a += pagg[(size_t)b * 64 + t];
    aggS[t] = a * invS;                // aggregated[d]
  }
  if (t < 128) {
    int uid = user_idx[0], iid = item_idx[0];
    ctxS[t] = (t < 64) ? user_emb[(size_t)uid * 64 + t]
                       : item_emb[(size_t)iid * 64 + (t - 64)];
  }
  __syncthreads();
  if (t < 32) {
    float hv = val_b1[t], ha = adv_b1[t];
    for (int d = 0; d < 64; ++d) {
      hv += val_W1[(size_t)t * 64 + d] * aggS[d];
      ha += adv_W1[(size_t)t * 64 + d] * aggS[d];
    }
    hidV[t] = fmaxf(hv, 0.0f);
    hidA[t] = fmaxf(ha, 0.0f);
    float hs = saf_b1[t];
    for (int k = 0; k < 128; ++k) hs += saf_W1[(size_t)t * 128 + k] * ctxS[k];
    hidS[t] = fmaxf(hs, 0.0f);
  }
  __syncthreads();
  if (t == 0) {
    float val = val_b2[0], adv = adv_b2[0], sz = saf_b2[0];
    for (int j = 0; j < 32; ++j) {
      val += hidV[j] * val_W2[j];
      adv += hidA[j] * adv_W2[j];
      sz  += hidS[j] * saf_W2[j];
    }
    float q = val + (adv - adv);       // advantage is [1] -> mean == advantage
    float saf = sigmoidf(sz);
    d_out[0] = q * saf;
  }
}

// ---------------- K6: normalize attention weights -----------------------
__global__ void k6_norm(float* outw, const float* finals) {
  int i = blockIdx.x * 256 + threadIdx.x;
  if (i < kP) outw[i] *= finals[2];
}

extern "C" void kernel_launch(void* const* d_in, const int* in_sizes, int n_in,
                              void* d_out, int out_size, void* d_ws, size_t ws_size,
                              hipStream_t stream) {
  const int*   user_idx   = (const int*)  d_in[0];
  const int*   item_idx   = (const int*)  d_in[1];
  const int*   node_ids   = (const int*)  d_in[2];
  const int*   node_types = (const int*)  d_in[3];
  const int*   rel_idx    = (const int*)  d_in[4];
  const float* user_emb   = (const float*)d_in[5];
  const float* item_emb   = (const float*)d_in[6];
  const float* rel_emb    = (const float*)d_in[7];
  const float* type_emb   = (const float*)d_in[8];
  const float* W_ih   = (const float*)d_in[9];
  const float* W_hh   = (const float*)d_in[10];
  const float* b_ih   = (const float*)d_in[11];
  const float* b_hh   = (const float*)d_in[12];
  const float* att_W1 = (const float*)d_in[13];
  const float* att_b1 = (const float*)d_in[14];
  const float* att_W2 = (const float*)d_in[15];
  const float* att_b2 = (const float*)d_in[16];
  const float* val_W1 = (const float*)d_in[17];
  const float* val_b1 = (const float*)d_in[18];
  const float* val_W2 = (const float*)d_in[19];
  const float* val_b2 = (const float*)d_in[20];
  const float* adv_W1 = (const float*)d_in[21];
  const float* adv_b1 = (const float*)d_in[22];
  const float* adv_W2 = (const float*)d_in[23];
  const float* adv_b2 = (const float*)d_in[24];
  const float* saf_W1 = (const float*)d_in[25];
  const float* saf_b1 = (const float*)d_in[26];
  const float* saf_W2 = (const float*)d_in[27];
  const float* saf_b2 = (const float*)d_in[28];

  float* ws     = (float*)d_ws;
  float* score  = ws + OFF_SCORE;
  float* pemb   = ws + OFF_PEMB;
  float* cvec   = ws + OFF_CVEC;
  float* pmax   = ws + OFF_PMAX;
  float* psum   = ws + OFF_PSUM;
  float* pagg   = ws + OFF_PAGG;
  float* finals = ws + OFF_FINAL;

  float* out      = (float*)d_out;   // [0]=prediction, [1..kP]=att_weights
  float* att_outw = out + 1;

  k0_ctx<<<1, 128, 0, stream>>>(user_idx, item_idx, user_emb, item_emb,
                                att_W1, att_b1, cvec);

  // dynamic LDS: W_ih(96K) + W_hh(32K) + attW1a(16K) + X(48K) + H(16K) + bias/cvec
  size_t shbytes = (size_t)(256 * 192 + 256 * 64 + 128 * 64 + 8 * 16 * 192 + 8 * 16 * 64) * 2
                 + (256 + 128) * 4;
  int nblk = (kP + 127) / 128;
  k1_lstm_att<<<nblk, 256, shbytes, stream>>>(
      node_ids, node_types, rel_idx, user_emb, item_emb, rel_emb, type_emb,
      W_ih, W_hh, b_ih, b_hh, att_W1, att_W2, att_b2, cvec, score, pemb);

  k2_blockmax<<<kNB2, 256, 0, stream>>>(score, pmax);
  k3_finalmax<<<1, 256, 0, stream>>>(pmax, finals);
  k4_expagg<<<kNB2, 256, 0, stream>>>(score, pemb, finals, att_outw, psum, pagg);
  k5_heads<<<1, 256, 0, stream>>>(psum, pagg, finals, user_idx, item_idx,
                                  user_emb, item_emb,
                                  val_W1, val_b1, val_W2, val_b2,
                                  adv_W1, adv_b1, adv_W2, adv_b2,
                                  saf_W1, saf_b1, saf_W2, saf_b2, out);
  k6_norm<<<(kP + 255) / 256, 256, 0, stream>>>(att_outw, finals);
}
